// MultiHeadAttention_28793460752493
// MI455X (gfx1250) — matrix operand order repaired and checked by reading, not compiled, not run on hardware
//
#include <hip/hip_runtime.h>
#include <hip/hip_bf16.h>
#include <cstdint>

// Problem constants (fixed by the reference)
#define BB 2
#define SS 2048
#define CC 1024
#define HH 16
#define DD 64
#define MM (BB*SS)   // 4096 rows in all GEMMs
#define RR 4

typedef __attribute__((ext_vector_type(16))) __bf16 v16bf;
typedef __attribute__((ext_vector_type(8)))  __bf16 v8bf;
typedef __attribute__((ext_vector_type(4)))  __bf16 v4bf;
typedef __attribute__((ext_vector_type(8)))  float  v8f;

__device__ __forceinline__ v16bf join8(v8bf lo, v8bf hi) {
  return __builtin_shufflevector(lo, hi, 0,1,2,3,4,5,6,7,8,9,10,11,12,13,14,15);
}

// A fragment (16x32 bf16, 16x16x32 WMMA):
// lane L: row = m0 + (L&15); hi = L>>4
// elements 0..7  -> k = k0 + hi*8 + i
// elements 8..15 -> k = k0 + 16 + hi*8 + (i-8)
__device__ __forceinline__ v16bf load_a_frag(const __bf16* A, int ldk, int m0, int k0) {
  const int lane = threadIdx.x & 31;
  const int row  = m0 + (lane & 15);
  const int hi   = lane >> 4;
  const __bf16* p = A + (size_t)row * ldk + k0 + hi * 8;
  v8bf lo = *(const v8bf*)p;
  v8bf hh = *(const v8bf*)(p + 16);
  return join8(lo, hh);
}

// B fragment (32x16 bf16): lane L: col n = n0 + (L&15); hi = L>>4
// elements 0..7 -> k = k0 + hi*8 + i; elements 8..15 -> k = k0 + 16 + hi*8 + (i-8).
__device__ __forceinline__ v16bf load_b_frag(const __bf16* Bm, int ldk, int n0, int k0) {
  const int lane = threadIdx.x & 31;
  const int col  = n0 + (lane & 15);
  const int hi   = lane >> 4;
  const __bf16* p = Bm + (size_t)col * ldk + k0 + hi * 8;
  v8bf lo = *(const v8bf*)p;
  v8bf hh = *(const v8bf*)(p + 16);
  return join8(lo, hh);
}

__device__ __forceinline__ v8f wmma_bf16(v16bf a, v16bf b, v8f c) {
  return __builtin_amdgcn_wmma_f32_16x16x32_bf16(false, a, false, b, (short)0, c, false, false);
}

// Async global->LDS copy (gfx1250 VGLOBAL async ops, tracked by ASYNCcnt).
// GVS mode: addr = SADDR(64) + VADDR(32, byte offset). vdst = LDS byte address.
__device__ __forceinline__ void async_load_b128(uint32_t lds_addr,
                                                unsigned long long gbase,
                                                uint32_t byte_off) {
  asm volatile("global_load_async_to_lds_b128 %0, %1, %2"
               :: "v"(lds_addr), "v"(byte_off), "s"(gbase)
               : "memory");
}
__device__ __forceinline__ void wait_async0() {
  asm volatile("s_wait_asynccnt 0" ::: "memory");
}
// generic shared pointer -> 32-bit LDS address (low 32 bits of flat shared addr)
__device__ __forceinline__ uint32_t lds_addr_of(const void* p) {
  return (uint32_t)(uintptr_t)p;
}

// cross-lane reductions over the 16-lane half-wave (rows 0-7 live in lanes 0-15,
// rows 8-15 in lanes 16-31, so xor masks 1,2,4,8 stay within the half).
template<int PAT>
__device__ __forceinline__ float dswz(float v) {
  return __int_as_float(__builtin_amdgcn_ds_swizzle(__float_as_int(v), PAT));
}
__device__ __forceinline__ float rmax16(float v) {
  v = fmaxf(v, dswz<0x041F>(v));  // SWAPX1
  v = fmaxf(v, dswz<0x081F>(v));  // SWAPX2
  v = fmaxf(v, dswz<0x101F>(v));  // SWAPX4
  v = fmaxf(v, dswz<0x201F>(v));  // SWAPX8
  return v;
}
__device__ __forceinline__ float rsum16(float v) {
  v += dswz<0x041F>(v);
  v += dswz<0x081F>(v);
  v += dswz<0x101F>(v);
  v += dswz<0x201F>(v);
  return v;
}

// ---------------------------------------------------------------- convert
__global__ void cvt_bf16_kernel(const float4* __restrict__ src,
                                v4bf* __restrict__ dst, int n4) {
  int i = blockIdx.x * blockDim.x + threadIdx.x;
  int stride = gridDim.x * blockDim.x;
  for (; i < n4; i += stride) {
    float4 f = src[i];
    v4bf o = {(__bf16)f.x, (__bf16)f.y, (__bf16)f.z, (__bf16)f.w};
    dst[i] = o;
  }
}

// ---------------------------------------------------------------- GEMM
// Y[m,n] = sum_k A[m,k] * W[n,k] + bias[n]
// mode 0: store bf16 at (B,H,S,D)   (Q / K)
// mode 1: store bf16 at (B,H,D,S)   (V transposed)
// mode 2: store fp32 row-major      (final output)
__global__ __launch_bounds__(256)
void gemm_bf16_kernel(const __bf16* __restrict__ A, const __bf16* __restrict__ W,
                      const float* __restrict__ bias, void* __restrict__ out, int mode)
{
  const int wave = threadIdx.x >> 5;
  const int m_wave = blockIdx.y * 128 + (wave >> 1) * 32;
  const int n_wave = blockIdx.x * 64  + (wave & 1) * 32;
  v8f acc[2][2] = {};

  for (int k0 = 0; k0 < CC; k0 += 32) {
    if (k0 + 32 < CC) {
      __builtin_prefetch(A + (size_t)(m_wave + (threadIdx.x & 31)) * CC + k0 + 32, 0, 1);
      __builtin_prefetch(W + (size_t)(n_wave + (threadIdx.x & 31)) * CC + k0 + 32, 0, 1);
    }
    v16bf a0 = load_a_frag(A, CC, m_wave,      k0);
    v16bf a1 = load_a_frag(A, CC, m_wave + 16, k0);
    v16bf b0 = load_b_frag(W, CC, n_wave,      k0);
    v16bf b1 = load_b_frag(W, CC, n_wave + 16, k0);
    acc[0][0] = wmma_bf16(a0, b0, acc[0][0]);
    acc[0][1] = wmma_bf16(a0, b1, acc[0][1]);
    acc[1][0] = wmma_bf16(a1, b0, acc[1][0]);
    acc[1][1] = wmma_bf16(a1, b1, acc[1][1]);
  }

  const int lane = threadIdx.x & 31;
  const int l15 = lane & 15, hi = lane >> 4;
  for (int i = 0; i < 2; ++i)
    for (int j = 0; j < 2; ++j) {
      const int n = n_wave + j * 16 + l15;
      const float bv = bias[n];
      #pragma unroll
      for (int r = 0; r < 8; ++r) {
        const int m = m_wave + i * 16 + r + 8 * hi;
        const float v = acc[i][j][r] + bv;
        if (mode == 2) {
          ((float*)out)[(size_t)m * CC + n] = v;
        } else {
          const int b = m >> 11, s = m & (SS - 1);     // S = 2048
          const int h = n >> 6,  d = n & (DD - 1);     // D = 64
          if (mode == 0)
            ((__bf16*)out)[(((size_t)(b * HH + h)) * SS + s) * DD + d] = (__bf16)v;
          else
            ((__bf16*)out)[(((size_t)(b * HH + h)) * DD + d) * SS + s] = (__bf16)v;
        }
      }
    }
}

// ---------------------------------------------------------------- rel-pos projection
// proj[bh, q, r] = inv * dot(Q[bh,q,:], rel_key[r,:])   (stride 12 for alignment)
__global__ void relproj_kernel(const __bf16* __restrict__ Q,
                               const float* __restrict__ rel_key,
                               float* __restrict__ proj)
{
  const int idx = blockIdx.x * blockDim.x + threadIdx.x;
  const int total = BB * HH * SS * 9;
  if (idx >= total) return;
  const int r = idx % 9;
  const int row = idx / 9;
  const __bf16* q = Q + (size_t)row * DD;
  const float* rk = rel_key + r * DD;
  float acc = 0.f;
  #pragma unroll
  for (int d = 0; d < DD; ++d) acc += (float)q[d] * rk[d];
  proj[(size_t)row * 12 + r] = acc * 0.125f;   // inv = 1/sqrt(64)
}

// ---------------------------------------------------------------- flash attention
// 1 block = 4 waves = 64 queries of one (b,h); 1 wave = 16 queries.
// K/V tiles for each 32-key block are staged into LDS with async global->LDS
// copies (double-buffered; copy of block i+1 overlaps compute of block i).
// All 4 waves consume the same LDS tiles (4x global-traffic reduction).
__global__ __launch_bounds__(128)
void attn_kernel(const __bf16* __restrict__ Q, const __bf16* __restrict__ K,
                 const __bf16* __restrict__ Vt, const float* __restrict__ proj,
                 const int* __restrict__ mask, __bf16* __restrict__ O)
{
  __shared__ __align__(128) __bf16 Kt[2][32][64];   // [buf][key][d]   4KB/buf
  __shared__ __align__(128) __bf16 Vtile[2][64][32];// [buf][d][key]   4KB/buf
  __shared__ __align__(64)  __bf16 ldsP[4][16][32];
  __shared__ float ldsProj[4][16][9];

  const int tid  = threadIdx.x;
  const int wave = tid >> 5;
  const int lane = tid & 31;
  const int l15 = lane & 15, hi = lane >> 4;
  const int bh = blockIdx.y;
  const int b = bh / HH, h = bh % HH;
  const int q0 = blockIdx.x * 64 + wave * 16;

  const __bf16* Qb = Q  + (size_t)bh * SS * DD;
  const __bf16* Kb = K  + (size_t)bh * SS * DD;
  const __bf16* Vb = Vt + (size_t)bh * DD * SS;
  const float*  Pj = proj + (size_t)bh * SS * 12;
  const int*    Mk = mask + (size_t)b * SS;

  const unsigned long long kbase = (unsigned long long)(uintptr_t)Kb;
  const unsigned long long vbase = (unsigned long long)(uintptr_t)Vb;
  const int vrow = tid >> 1, vseg = tid & 1;

  // stage one 32-key block (K: 4KB contiguous; V: 64 rows x 64B) into buffer `buf`
  auto stage = [&](int buf, int kb) {
    const uint32_t ldsK = lds_addr_of(&Kt[buf][0][0]);
    const uint32_t gK   = (uint32_t)(kb * DD * 2);
    async_load_b128(ldsK + tid * 16,        kbase, gK + tid * 16);
    async_load_b128(ldsK + 2048 + tid * 16, kbase, gK + 2048 + tid * 16);
    const uint32_t ldsV = lds_addr_of(&Vtile[buf][vrow][0]) + vseg * 32;
    const uint32_t gV   = (uint32_t)((vrow * SS + kb) * 2) + vseg * 32;
    async_load_b128(ldsV,      vbase, gV);
    async_load_b128(ldsV + 16, vbase, gV + 16);
  };

  stage(0, 0);

  if (lane < 16) {
    #pragma unroll
    for (int r = 0; r < 9; ++r)
      ldsProj[wave][lane][r] = Pj[(size_t)(q0 + lane) * 12 + r];
  }

  const v16bf qa0 = load_a_frag(Qb, DD, q0, 0);
  const v16bf qa1 = load_a_frag(Qb, DD, q0, 32);

  v8f o[4] = {};
  float m8[8], l8[8];
  #pragma unroll
  for (int r = 0; r < 8; ++r) { m8[r] = -1e30f; l8[r] = 0.f; }

  wait_async0();
  __syncthreads();

  for (int kb = 0; kb < SS; kb += 32) {
    const int buf = (kb >> 5) & 1;
    if (kb + 32 < SS) stage(buf ^ 1, kb + 32);

    const __bf16* Ksh = &Kt[buf][0][0];
    const __bf16* Vsh = &Vtile[buf][0][0];

    // scores: 16 queries x 32 keys (fragments read from LDS, shared by 4 waves)
    v8f s0 = {}, s1 = {};
    s0 = wmma_bf16(qa0, load_b_frag(Ksh, DD, 0,  0),  s0);
    s0 = wmma_bf16(qa1, load_b_frag(Ksh, DD, 0,  32), s0);
    s1 = wmma_bf16(qa0, load_b_frag(Ksh, DD, 16, 0),  s1);
    s1 = wmma_bf16(qa1, load_b_frag(Ksh, DD, 16, 32), s1);

    const float add0 = Mk[kb + l15]      ? 0.f : -1e9f;
    const float add1 = Mk[kb + 16 + l15] ? 0.f : -1e9f;
    const float inv = 0.125f;

    float p0[8], p1[8], scale[8];
    #pragma unroll
    for (int r = 0; r < 8; ++r) {
      const int row = r + 8 * hi;
      const int qi = q0 + row;
      int d0 = (kb + l15) - qi;       d0 = d0 < -RR ? -RR : (d0 > RR ? RR : d0);
      int d1 = (kb + 16 + l15) - qi;  d1 = d1 < -RR ? -RR : (d1 > RR ? RR : d1);
      const float v0 = s0[r] * inv + ldsProj[wave][row][d0 + RR] + add0;
      const float v1 = s1[r] * inv + ldsProj[wave][row][d1 + RR] + add1;
      const float rmax = rmax16(fmaxf(v0, v1));
      const float nm = fmaxf(m8[r], rmax);
      scale[r] = __expf(m8[r] - nm);
      p0[r] = __expf(v0 - nm);
      p1[r] = __expf(v1 - nm);
      l8[r] = l8[r] * scale[r] + rsum16(p0[r] + p1[r]);
      m8[r] = nm;
    }
    #pragma unroll
    for (int j = 0; j < 4; ++j)
      #pragma unroll
      for (int r = 0; r < 8; ++r)
        o[j][r] *= scale[r];

    // stage P (C-layout -> LDS row-major bf16); same-wave LDS ordering only
    #pragma unroll
    for (int r = 0; r < 8; ++r) {
      const int row = r + 8 * hi;
      ldsP[wave][row][l15]      = (__bf16)p0[r];
      ldsP[wave][row][l15 + 16] = (__bf16)p1[r];
    }
    const v8bf plo = *(const v8bf*)&ldsP[wave][l15][hi * 8];
    const v8bf phh = *(const v8bf*)&ldsP[wave][l15][16 + hi * 8];
    const v16bf pa = join8(plo, phh);

    // PV: B fragments from LDS V tile ([d][key], row stride 32 elements)
    #pragma unroll
    for (int j = 0; j < 4; ++j)
      o[j] = wmma_bf16(pa, load_b_frag(Vsh, 32, j * 16, 0), o[j]);

    wait_async0();       // next buffer arrived
    __syncthreads();     // all waves done reading current buffer
  }

  #pragma unroll
  for (int j = 0; j < 4; ++j) {
    #pragma unroll
    for (int r = 0; r < 8; ++r) {
      const int row = r + 8 * hi;
      const int qi = q0 + row;
      const float v = o[j][r] / l8[r];
      O[((size_t)(b * SS + qi)) * CC + h * DD + j * 16 + l15] = (__bf16)v;
    }
  }
}

// ---------------------------------------------------------------- host
extern "C" void kernel_launch(void* const* d_in, const int* in_sizes, int n_in,
                              void* d_out, int out_size, void* d_ws, size_t ws_size,
                              hipStream_t stream)
{
  const float* x    = (const float*)d_in[0];
  const int*   mask = (const int*)  d_in[1];
  const float* Wq   = (const float*)d_in[2];
  const float* bq   = (const float*)d_in[3];
  const float* Wk   = (const float*)d_in[4];
  const float* bk   = (const float*)d_in[5];
  const float* Wv   = (const float*)d_in[6];
  const float* bv   = (const float*)d_in[7];
  const float* Wo   = (const float*)d_in[8];
  const float* bo   = (const float*)d_in[9];
  const float* rel  = (const float*)d_in[10];

  char* ws = (char*)d_ws;
  size_t off = 0;
  auto alloc = [&](size_t bytes) -> char* {
    char* p = ws + off;
    off += (bytes + 255) & ~(size_t)255;
    return p;
  };
  __bf16* xbf  = (__bf16*)alloc((size_t)MM * CC * 2);
  __bf16* Wqb  = (__bf16*)alloc((size_t)CC * CC * 2);
  __bf16* Wkb  = (__bf16*)alloc((size_t)CC * CC * 2);
  __bf16* Wvb  = (__bf16*)alloc((size_t)CC * CC * 2);
  __bf16* Wob  = (__bf16*)alloc((size_t)CC * CC * 2);
  __bf16* Qb   = (__bf16*)alloc((size_t)MM * CC * 2);   // (B,H,S,D)
  __bf16* Kbf  = (__bf16*)alloc((size_t)MM * CC * 2);   // (B,H,S,D)
  __bf16* Vtb  = (__bf16*)alloc((size_t)MM * CC * 2);   // (B,H,D,S)
  float*  pjf  = (float*) alloc((size_t)BB * HH * SS * 12 * 4);
  __bf16* Ab   = (__bf16*)alloc((size_t)MM * CC * 2);   // attention out (B,S,C)

  // fp32 -> bf16 conversions
  {
    const int nX = MM * CC / 4, nW = CC * CC / 4;
    cvt_bf16_kernel<<<dim3(1024), dim3(256), 0, stream>>>((const float4*)x,  (v4bf*)xbf, nX);
    cvt_bf16_kernel<<<dim3(512),  dim3(256), 0, stream>>>((const float4*)Wq, (v4bf*)Wqb, nW);
    cvt_bf16_kernel<<<dim3(512),  dim3(256), 0, stream>>>((const float4*)Wk, (v4bf*)Wkb, nW);
    cvt_bf16_kernel<<<dim3(512),  dim3(256), 0, stream>>>((const float4*)Wv, (v4bf*)Wvb, nW);
    cvt_bf16_kernel<<<dim3(512),  dim3(256), 0, stream>>>((const float4*)Wo, (v4bf*)Wob, nW);
  }

  const dim3 gg(CC / 64, MM / 128);   // (16, 32)
  gemm_bf16_kernel<<<gg, 256, 0, stream>>>(xbf, Wqb, bq, Qb,  0);
  gemm_bf16_kernel<<<gg, 256, 0, stream>>>(xbf, Wkb, bk, Kbf, 0);
  gemm_bf16_kernel<<<gg, 256, 0, stream>>>(xbf, Wvb, bv, Vtb, 1);

  relproj_kernel<<<dim3((BB * HH * SS * 9 + 255) / 256), dim3(256), 0, stream>>>(Qb, rel, pjf);

  attn_kernel<<<dim3(SS / 64, BB * HH), dim3(128), 0, stream>>>(Qb, Kbf, Vtb, pjf, mask, Ab);

  gemm_bf16_kernel<<<gg, 256, 0, stream>>>(Ab, Wob, bo, d_out, 2);
}
